// SwinTransformerBlock_45552423141400
// MI455X (gfx1250) — compile-verified, matrix operand order checked by reading, not run on hardware
//
#include <hip/hip_runtime.h>
#include <hip/hip_bf16.h>

typedef __attribute__((ext_vector_type(16))) _Float16 v16h;
typedef __attribute__((ext_vector_type(8)))  float    v8f;

#define H_      128
#define WS_     8
#define SHIFT_  4
#define C_      128
#define HEADS_  4
#define HD_     32
#define T_      2
#define N_      64
#define TN_     128

// ---------------------------------------------------------------------------
// WMMA helpers (CDNA5 wave32, 16x16x32 f16 -> f32)
// ---------------------------------------------------------------------------
__device__ __forceinline__ v8f wmma_f16(v16h a, v16h b, v8f c) {
  // 8 args: (neg_a, A, neg_b, B, c_mod, C, reuse_a, reuse_b)
  return __builtin_amdgcn_wmma_f32_16x16x32_f16(false, a, false, b, (short)0, c,
                                                false, false);
}

// A-operand (16xK tile, row-major f16 in LDS). Also used for B when the
// B-operand is iterated row-major over K (i.e. computing X @ W^T).
// Lane mapping: M = lane&15; lanes 16-31 take K+8 within each 16-K block.
__device__ __forceinline__ v16h fragA_lds(const _Float16* base, int m0, int ld,
                                          int k0, int lane) {
  const _Float16* p = base + (size_t)(m0 + (lane & 15)) * ld + k0 + ((lane >> 4) << 3);
  v16h a;
#pragma unroll
  for (int e = 0; e < 8; ++e)  a[e] = p[e];
#pragma unroll
  for (int e = 8; e < 16; ++e) a[e] = p[e + 8];
  return a;
}

// B-operand from global f32 weights stored (N, K) row-major: y = x @ W^T.
__device__ __forceinline__ v16h fragB_rm_g(const float* base, int n0, int ld,
                                           int k0, int lane) {
  const float* p = base + (size_t)(n0 + (lane & 15)) * ld + k0 + ((lane >> 4) << 3);
  v16h b;
#pragma unroll
  for (int e = 0; e < 8; ++e)  b[e] = (_Float16)p[e];
#pragma unroll
  for (int e = 8; e < 16; ++e) b[e] = (_Float16)p[e + 8];
  return b;
}

// B-operand from LDS stored (K, N) row-major (column-major over K per lane):
// used for P @ V where V is [token(K)][hd(N)].
__device__ __forceinline__ v16h fragB_cm_lds(const _Float16* base, int n0, int ld,
                                             int k0, int lane) {
  const _Float16* p = base + n0 + (lane & 15) + (size_t)(k0 + ((lane >> 4) << 3)) * ld;
  v16h b;
#pragma unroll
  for (int e = 0; e < 8; ++e)  b[e] = p[(size_t)e * ld];
#pragma unroll
  for (int e = 8; e < 16; ++e) b[e] = p[(size_t)(e + 8) * ld];
  return b;
}

// Async bulk copy: 16B per lane, global -> LDS, no VGPR data round-trip.
// Tracked by ASYNCcnt (CDNA5 ISA).
__device__ __forceinline__ void async_g2l_b128(void* lds_dst, const void* gsrc) {
  unsigned ldsa = (unsigned)(size_t)lds_dst;   // flat->LDS: low 32 bits
  asm volatile("global_load_async_to_lds_b128 %0, %1, off"
               :: "v"(ldsa), "v"(gsrc) : "memory");
}
__device__ __forceinline__ void wait_async0() {
  asm volatile("s_wait_asynccnt 0x0" ::: "memory");
}

// ---------------------------------------------------------------------------
// Kernel 1: shifted-window gather + QKV projection (WMMA), q pre-scaled.
// One workgroup per window-batch b_ (B*nW = 1024), 256 threads = 8 waves.
// ---------------------------------------------------------------------------
__global__ void __launch_bounds__(256) qkv_kernel(
    const float* __restrict__ x_v, const float* __restrict__ qkv_w,
    const float* __restrict__ qkv_b, _Float16* __restrict__ qw,
    _Float16* __restrict__ kw, _Float16* __restrict__ vw) {
  extern __shared__ char smem[];
  _Float16* xs = (_Float16*)smem;   // 128 x 130 (token x channel, f16)
  _Float16* wl = xs + 128 * 130;    // 384 x 130 (qkv_w staged f16)

  const int tid  = threadIdx.x;
  const int b_   = blockIdx.x;
  const int b    = b_ >> 8;     // / nW (256)
  const int wIdx = b_ & 255;
  const int wh   = wIdx >> 4;
  const int ww   = wIdx & 15;

  // stage weights f32 -> f16
  for (int i = tid; i < 384 * 128; i += 256) {
    int rr = i >> 7, cc = i & 127;
    wl[rr * 130 + cc] = (_Float16)qkv_w[i];
  }
  // stage shifted window tile: token (t, n=(r,c)) <- x_v[(y+4)%128, (x+4)%128]
  {
    int tok = tid >> 1, half = tid & 1;
    int t = tok >> 6, n = tok & 63;
    int r = n >> 3, c = n & 7;
    int hs  = (wh * WS_ + r + SHIFT_) & 127;
    int ws2 = (ww * WS_ + c + SHIFT_) & 127;
    const float* src = x_v + ((((size_t)b * T_ + t) * (H_ * H_) + hs * H_ + ws2) << 7) + half * 64;
    _Float16* dst = xs + tok * 130 + half * 64;
#pragma unroll
    for (int i = 0; i < 64; i += 4) {
      float4 v = *(const float4*)(src + i);
      dst[i] = (_Float16)v.x; dst[i + 1] = (_Float16)v.y;
      dst[i + 2] = (_Float16)v.z; dst[i + 3] = (_Float16)v.w;
    }
  }
  __syncthreads();

  const int w = tid >> 5, lane = tid & 31;
  const int m0 = w * 16;
  const int hi = (lane >> 4) << 3;
  const size_t bh_base = (size_t)b_ * HEADS_;

  for (int nt = 0; nt < 24; ++nt) {           // 384 output cols
    v8f acc = {};
#pragma unroll
    for (int kt = 0; kt < 4; ++kt) {          // K = 128
      v16h a   = fragA_lds(xs, m0, 130, kt * 32, lane);
      v16h bfr = fragA_lds(wl, nt * 16, 130, kt * 32, lane);
      acc = wmma_f16(a, bfr, acc);
    }
    int col = nt * 16 + (lane & 15);
    float bias  = qkv_b[col];
    float scale = (col < 128) ? 0.17677669529663687f : 1.0f;  // hd^-0.5 for q
    int sel = col >> 7, c0 = col & 127;
    int h = c0 >> 5, d = c0 & 31;
    _Float16* dstb = (sel == 0) ? qw : (sel == 1) ? kw : vw;
#pragma unroll
    for (int r = 0; r < 8; ++r) {
      int row = m0 + r + hi;
      float val = (acc[r] + bias) * scale;
      dstb[((bh_base + h) * TN_ + row) * HD_ + d] = (_Float16)val;
    }
  }
}

// ---------------------------------------------------------------------------
// Kernel 2: windowed attention (QK^T + rel-pos bias + shift mask + softmax +
// P@V), output projection, shortcut add, reverse-roll scatter.
// q/k/v staged with GLOBAL_LOAD_ASYNC_TO_LDS_B128 (ASYNCcnt path).
// ---------------------------------------------------------------------------
__device__ __forceinline__ int region3(int y) {
  return y < (H_ - WS_) ? 0 : (y < (H_ - SHIFT_) ? 1 : 2);
}

__global__ void __launch_bounds__(256) attn_kernel(
    const _Float16* __restrict__ qw, const _Float16* __restrict__ kw,
    const _Float16* __restrict__ vw, const float* __restrict__ rpb_table,
    const float* __restrict__ proj_w, const float* __restrict__ proj_b,
    const float* __restrict__ x_v, float* __restrict__ xres) {
  extern __shared__ char smem[];
  _Float16* qs = (_Float16*)smem;        // 4 heads x 128 x 32
  _Float16* ks = qs + 16384;
  _Float16* vs = ks + 16384;
  _Float16* attn_s = vs + 16384;         // 128 x 130 probs (f16)
  _Float16* ctx    = attn_s + 128 * 130; // 128 x 130 context (all heads)

  const int tid  = threadIdx.x;
  const int b_   = blockIdx.x;
  const int b    = b_ >> 8;
  const int wIdx = b_ & 255;
  const int wh   = wIdx >> 4;
  const int ww   = wIdx & 15;
  const size_t base = (size_t)b_ * 16384;

  // async global -> LDS staging, 16B per lane per issue, no VGPR data
  for (int i = tid * 8; i < 16384; i += 2048) {
    async_g2l_b128(qs + i, qw + base + i);
    async_g2l_b128(ks + i, kw + base + i);
    async_g2l_b128(vs + i, vw + base + i);
  }
  wait_async0();
  __syncthreads();

  const int w = tid >> 5, lane = tid & 31;
  const int m0  = w * 16;
  const int hi  = (lane >> 4) << 3;
  const int nlo = lane & 15;

  for (int h = 0; h < HEADS_; ++h) {
    const _Float16* qh = qs + h * 4096;
    const _Float16* kh = ks + h * 4096;
    const _Float16* vh = vs + h * 4096;

    // attn tiles: this wave owns q rows m0..m0+15 over all 128 k-tokens.
    v16h aq = fragA_lds(qh, m0, HD_, 0, lane);   // K = hd = 32 -> one WMMA/tile
    v8f acc[8];
#pragma unroll
    for (int nt = 0; nt < 8; ++nt) {
      v16h bf = fragA_lds(kh, nt * 16, HD_, 0, lane);
      v8f z = {};
      acc[nt] = wmma_f16(aq, bf, z);
    }
    // bias + mask + row softmax (rows wave-private; 16-lane shuffle reduce)
#pragma unroll
    for (int r = 0; r < 8; ++r) {
      int row = m0 + r + hi;
      int qn = row & 63;
      int r1 = qn >> 3, c1 = qn & 7;
      int regq = region3(wh * WS_ + r1) * 3 + region3(ww * WS_ + c1);
#pragma unroll
      for (int nt = 0; nt < 8; ++nt) {
        int ktok = nt * 16 + nlo;
        int kn = ktok & 63;
        int r2 = kn >> 3, c2 = kn & 7;
        int regk = region3(wh * WS_ + r2) * 3 + region3(ww * WS_ + c2);
        int ridx = (r1 - r2 + 7) * 15 + (c1 - c2 + 7);
        float bias = rpb_table[ridx * HEADS_ + h];
        if (regq != regk) bias -= 100.0f;
        acc[nt][r] += bias;
      }
      float mx = -3.0e38f;
#pragma unroll
      for (int nt = 0; nt < 8; ++nt) mx = fmaxf(mx, acc[nt][r]);
#pragma unroll
      for (int off = 1; off < 16; off <<= 1) mx = fmaxf(mx, __shfl_xor(mx, off, 32));
      float sum = 0.0f;
#pragma unroll
      for (int nt = 0; nt < 8; ++nt) {
        float e = __expf(acc[nt][r] - mx);
        acc[nt][r] = e; sum += e;
      }
#pragma unroll
      for (int off = 1; off < 16; off <<= 1) sum += __shfl_xor(sum, off, 32);
      float inv = 1.0f / sum;
#pragma unroll
      for (int nt = 0; nt < 8; ++nt)
        attn_s[(size_t)row * 130 + nt * 16 + nlo] = (_Float16)(acc[nt][r] * inv);
    }
    // context = P @ V (K = 128, N = 32); rows wave-private -> no barrier
#pragma unroll
    for (int nt2 = 0; nt2 < 2; ++nt2) {
      v8f oacc = {};
#pragma unroll
      for (int kt = 0; kt < 4; ++kt) {
        v16h a  = fragA_lds(attn_s, m0, 130, kt * 32, lane);
        v16h bf = fragB_cm_lds(vh, nt2 * 16, HD_, kt * 32, lane);
        oacc = wmma_f16(a, bf, oacc);
      }
      int col = h * HD_ + nt2 * 16 + nlo;
#pragma unroll
      for (int r = 0; r < 8; ++r)
        ctx[(size_t)(m0 + r + hi) * 130 + col] = (_Float16)oacc[r];
    }
  }

  // projection + bias + shortcut, reverse-roll scatter (== gather index)
  for (int nt = 0; nt < 8; ++nt) {
    v8f acc = {};
#pragma unroll
    for (int kt = 0; kt < 4; ++kt) {
      v16h a  = fragA_lds(ctx, m0, 130, kt * 32, lane);
      v16h bf = fragB_rm_g(proj_w, nt * 16, C_, kt * 32, lane);
      acc = wmma_f16(a, bf, acc);
    }
    int col = nt * 16 + nlo;
    float pb = proj_b[col];
#pragma unroll
    for (int r = 0; r < 8; ++r) {
      int row = m0 + r + hi;
      int t = row >> 6, n = row & 63;
      int rr = n >> 3, cc = n & 7;
      int hs  = (wh * WS_ + rr + SHIFT_) & 127;
      int ws2 = (ww * WS_ + cc + SHIFT_) & 127;
      size_t idx = ((((size_t)b * T_ + t) * (H_ * H_) + hs * H_ + ws2) << 7) + col;
      xres[idx] = acc[r] + pb + x_v[idx];
    }
  }
}

// ---------------------------------------------------------------------------
// Kernel 3: LN2 -> fc1(+GELU) -> fc2 -> residual -> LN1 -> out.
// 64-token tiles, 2048 workgroups, 256 threads = 8 waves.
// ---------------------------------------------------------------------------
__global__ void __launch_bounds__(256) mlp_kernel(
    const float* __restrict__ xres,
    const float* __restrict__ n2w, const float* __restrict__ n2b,
    const float* __restrict__ fc1w, const float* __restrict__ fc1b,
    const float* __restrict__ fc2w, const float* __restrict__ fc2b,
    const float* __restrict__ n1w, const float* __restrict__ n1b,
    float* __restrict__ out) {
  extern __shared__ char smem[];
  float*    xs    = (float*)smem;                  // 64 x 132 (f32 residual)
  _Float16* ln_s  = (_Float16*)(xs + 64 * 132);    // 64 x 130 (LN2 f16)
  _Float16* hid_s = ln_s + 64 * 130;               // 64 x 520 (gelu hidden f16)

  const int tid = threadIdx.x;
  const size_t tok0 = (size_t)blockIdx.x * 64;

  // pre-touch MLP weights into GL2/WGP$ (global_prefetch_b8)
  __builtin_prefetch(fc1w + (size_t)tid * 256, 0, 1);
  __builtin_prefetch(fc2w + (size_t)tid * 256, 0, 1);

  for (int i = tid * 4; i < 64 * 128; i += 1024) {
    float4 v = *(const float4*)(xres + tok0 * C_ + i);
    int rr = i >> 7, cc = i & 127;
    *(float4*)(xs + rr * 132 + cc) = v;
  }
  __syncthreads();

  // LN2 (4 lanes per token, shuffle-combine partials)
  {
    int t_ = tid >> 2, part = tid & 3;
    const float* px = xs + t_ * 132 + part * 32;
    float s = 0.f, sq = 0.f;
#pragma unroll
    for (int j = 0; j < 32; ++j) { float v = px[j]; s += v; sq += v * v; }
    s  += __shfl_xor(s, 1, 32);  s  += __shfl_xor(s, 2, 32);
    sq += __shfl_xor(sq, 1, 32); sq += __shfl_xor(sq, 2, 32);
    float mu  = s * (1.0f / 128.0f);
    float var = sq * (1.0f / 128.0f) - mu * mu;
    float rs  = rsqrtf(var + 1e-5f);
    _Float16* pl = ln_s + t_ * 130 + part * 32;
#pragma unroll
    for (int j = 0; j < 32; ++j) {
      int col = part * 32 + j;
      pl[j] = (_Float16)((px[j] - mu) * rs * n2w[col] + n2b[col]);
    }
  }
  __syncthreads();

  const int w = tid >> 5, lane = tid & 31;
  const int hi  = (lane >> 4) << 3;
  const int nlo = lane & 15;

  // fc1 (64x128 @ 128x512) + exact GELU
  {
    int mt = w & 3, ng = w >> 2;
    int m0 = mt * 16;
    for (int j = 0; j < 16; ++j) {
      int n0 = (ng * 16 + j) * 16;
      v8f acc = {};
#pragma unroll
      for (int kt = 0; kt < 4; ++kt) {
        v16h a  = fragA_lds(ln_s, m0, 130, kt * 32, lane);
        v16h bf = fragB_rm_g(fc1w, n0, C_, kt * 32, lane);
        acc = wmma_f16(a, bf, acc);
      }
      int col = n0 + nlo;
      float bias = fc1b[col];
#pragma unroll
      for (int r = 0; r < 8; ++r) {
        float v = acc[r] + bias;
        float g = 0.5f * v * (1.0f + erff(v * 0.70710678118f));
        hid_s[(size_t)(m0 + r + hi) * 520 + col] = (_Float16)g;
      }
    }
  }
  __syncthreads();

  // fc2 (64x512 @ 512x128) + residual back into xs
  {
    int mt = w >> 1, nq = w & 1;
    int m0 = mt * 16;
    for (int j = 0; j < 4; ++j) {
      int n0 = (nq * 4 + j) * 16;
      v8f acc = {};
#pragma unroll
      for (int kt = 0; kt < 16; ++kt) {
        v16h a  = fragA_lds(hid_s, m0, 520, kt * 32, lane);
        v16h bf = fragB_rm_g(fc2w, n0, 512, kt * 32, lane);
        acc = wmma_f16(a, bf, acc);
      }
      int col = n0 + nlo;
      float bias = fc2b[col];
#pragma unroll
      for (int r = 0; r < 8; ++r) {
        int row = m0 + r + hi;
        xs[row * 132 + col] += acc[r] + bias;
      }
    }
  }
  __syncthreads();

  // LN1 -> output
  {
    int t_ = tid >> 2, part = tid & 3;
    const float* px = xs + t_ * 132 + part * 32;
    float s = 0.f, sq = 0.f;
#pragma unroll
    for (int j = 0; j < 32; ++j) { float v = px[j]; s += v; sq += v * v; }
    s  += __shfl_xor(s, 1, 32);  s  += __shfl_xor(s, 2, 32);
    sq += __shfl_xor(sq, 1, 32); sq += __shfl_xor(sq, 2, 32);
    float mu  = s * (1.0f / 128.0f);
    float var = sq * (1.0f / 128.0f) - mu * mu;
    float rs  = rsqrtf(var + 1e-5f);
    float* po = out + (tok0 + t_) * C_ + part * 32;
#pragma unroll
    for (int j = 0; j < 32; ++j) {
      int col = part * 32 + j;
      po[j] = (px[j] - mu) * rs * n1w[col] + n1b[col];
    }
  }
}

// ---------------------------------------------------------------------------
extern "C" void kernel_launch(void* const* d_in, const int* in_sizes, int n_in,
                              void* d_out, int out_size, void* d_ws, size_t ws_size,
                              hipStream_t stream) {
  const float* x_v   = (const float*)d_in[0];
  const float* qkvw  = (const float*)d_in[1];
  const float* qkvb  = (const float*)d_in[2];
  const float* projw = (const float*)d_in[3];
  const float* projb = (const float*)d_in[4];
  const float* rpb   = (const float*)d_in[5];
  const float* n1w   = (const float*)d_in[6];
  const float* n1b   = (const float*)d_in[7];
  const float* n2w   = (const float*)d_in[8];
  const float* n2b   = (const float*)d_in[9];
  const float* fc1w  = (const float*)d_in[10];
  const float* fc1b  = (const float*)d_in[11];
  const float* fc2w  = (const float*)d_in[12];
  const float* fc2b  = (const float*)d_in[13];
  float* out = (float*)d_out;

  const size_t QSZ = (size_t)1024 * HEADS_ * TN_ * HD_;  // halves per q/k/v
  _Float16* qw = (_Float16*)d_ws;
  _Float16* kw = qw + QSZ;
  _Float16* vw = kw + QSZ;
  float* xres = (float*)(vw + QSZ);

  size_t lds1 = (size_t)(128 * 130 + 384 * 130) * sizeof(_Float16);           // ~130 KB
  size_t lds2 = (size_t)(3 * 16384 + 2 * 128 * 130) * sizeof(_Float16);       // ~161 KB
  size_t lds3 = (size_t)64 * 132 * sizeof(float) +
                (size_t)(64 * 130 + 64 * 520) * sizeof(_Float16);             // ~114 KB

  qkv_kernel<<<1024, 256, lds1, stream>>>(x_v, qkvw, qkvb, qw, kw, vw);
  attn_kernel<<<1024, 256, lds2, stream>>>(qw, kw, vw, rpb, projw, projb, x_v, xres);
  mlp_kernel<<<2048, 256, lds3, stream>>>(xres, n2w, n2b, fc1w, fc1b,
                                          fc2w, fc2b, n1w, n1b, out);
}
